// MultiheadSelfAttention_54924041781999
// MI455X (gfx1250) — matrix operand order repaired and checked by reading, compiled, not run on hardware
//
#include <hip/hip_runtime.h>
#include <hip/hip_bf16.h>
#include <math.h>

#define SEQ   4096
#define DM    1024
#define NH    16
#define HD    64
#define QKV_N (3 * DM)

typedef __attribute__((ext_vector_type(16))) __bf16 v16bf;
typedef __attribute__((ext_vector_type(8)))  __bf16 v8bf;
typedef __attribute__((ext_vector_type(8)))  float  v8f;

// ---------------------------------------------------------------------------
// Fragment loaders for V_WMMA_F32_16X16X32_BF16 (wave32; contiguous k order, same for A and B)
// A (16x32, row-major MxK source): lane L -> row L&15, K = kb..kb+15,
// kb = 16*(L>>4). Two contiguous 16B loads.
// B (32x16, source stored NxK row-major): lane L -> col L&15, K = 16*(L>>4)..+15.
// One contiguous 32B load.
// ---------------------------------------------------------------------------
__device__ __forceinline__ v16bf load_a_frag(const __bf16* base, int lda, int lane) {
  const int row = lane & 15;
  const int kb  = (lane >> 4) << 4;
  const __bf16* p = base + (size_t)row * lda + kb;
  v8bf lo = *(const v8bf*)(p);
  v8bf hi = *(const v8bf*)(p + 8);
  v16bf r;
#pragma unroll
  for (int i = 0; i < 8; ++i) { r[i] = lo[i]; r[i + 8] = hi[i]; }
  return r;
}

__device__ __forceinline__ v16bf load_b_frag(const __bf16* base, int ldb, int lane) {
  const int col = lane & 15;
  const int kb  = (lane >> 4) << 4;
  return *(const v16bf*)(base + (size_t)col * ldb + kb);
}

// CDNA5 async global->LDS copy (16B per lane), tracked by ASYNCcnt.
// lds_off = workgroup-relative LDS byte address; gaddr = 64-bit global address.
__device__ __forceinline__ void async_copy_b128(unsigned lds_off, const __bf16* gptr) {
  unsigned long long ga = (unsigned long long)gptr;
  asm volatile("global_load_async_to_lds_b128 %0, %1, off"
               :: "v"(lds_off), "v"(ga) : "memory");
}

__device__ __forceinline__ void wait_asynccnt0() {
  asm volatile("s_wait_asynccnt 0x0" ::: "memory");
}

// ---------------------------------------------------------------------------
// fp32 -> bf16 conversion
// ---------------------------------------------------------------------------
__global__ void f32_to_bf16(const float* __restrict__ in, __bf16* __restrict__ out, int n) {
  int i = blockIdx.x * blockDim.x + threadIdx.x;
  if (i < n) out[i] = (__bf16)in[i];
}

// ---------------------------------------------------------------------------
// C[M,N] = A[M,K] * W[N,K]^T, bf16 inputs, f32 accumulate.
// One wave owns a 64x64 tile (4x4 WMMA frags); 4 independent waves per block.
// ---------------------------------------------------------------------------
template <bool OUT_F32>
__global__ void __launch_bounds__(128)
gemm_bf16(const __bf16* __restrict__ A, const __bf16* __restrict__ W,
          void* __restrict__ Cout, int M, int N, int K) {
  const int lane = threadIdx.x & 31;
  const int wave = threadIdx.x >> 5;
  const int wid  = blockIdx.x * 4 + wave;
  const int ntn  = N >> 6;
  const int mt   = (wid / ntn) << 6;
  const int nt   = (wid % ntn) << 6;
  if (mt >= M) return;

  v8f acc[4][4] = {};

  for (int k0 = 0; k0 < K; k0 += 32) {
    if (k0 + 64 < K) {  // pull next k-slab toward L2/L0 (global_prefetch_b8)
      __builtin_prefetch(A + (size_t)mt * K + k0 + 64, 0, 0);
      __builtin_prefetch(W + (size_t)nt * K + k0 + 64, 0, 0);
    }
    v16bf a[4], b[4];
#pragma unroll
    for (int i = 0; i < 4; ++i)
      a[i] = load_a_frag(A + (size_t)(mt + 16 * i) * K + k0, K, lane);
#pragma unroll
    for (int j = 0; j < 4; ++j)
      b[j] = load_b_frag(W + (size_t)(nt + 16 * j) * K + k0, K, lane);
#pragma unroll
    for (int i = 0; i < 4; ++i)
#pragma unroll
      for (int j = 0; j < 4; ++j)
        acc[i][j] = __builtin_amdgcn_wmma_f32_16x16x32_bf16(
            false, a[i], false, b[j], (short)0, acc[i][j], false, false);
  }

  const int halfsel = lane >> 4, col = lane & 15;
#pragma unroll
  for (int i = 0; i < 4; ++i)
#pragma unroll
    for (int j = 0; j < 4; ++j) {
      const int n = nt + 16 * j + col;
#pragma unroll
      for (int r = 0; r < 8; ++r) {
        const int m = mt + 16 * i + r + 8 * halfsel;
        if (OUT_F32) ((float*)Cout)[(size_t)m * N + n] = acc[i][j][r];
        else ((__bf16*)Cout)[(size_t)m * N + n] = (__bf16)acc[i][j][r];
      }
    }
}

// ---------------------------------------------------------------------------
// RoPE on Q and K + split to per-head [h][s][64] layout (bf16).
// ---------------------------------------------------------------------------
__global__ void rope_split(const __bf16* __restrict__ qkv, const int* __restrict__ pos,
                           __bf16* __restrict__ qh, __bf16* __restrict__ kh) {
  int idx = blockIdx.x * blockDim.x + threadIdx.x;
  if (idx >= SEQ * NH * (HD / 2)) return;
  const int i = idx & 31;
  const int h = (idx >> 5) & 15;
  const int s = idx >> 9;

  const float p   = (float)pos[s];
  const float inv = __powf(10000.0f, -(float)(2 * i) * (1.0f / (float)HD));
  const float ang = p * inv;
  const float c = __cosf(ang), sn = __sinf(ang);

  const size_t qb = (size_t)s * QKV_N + h * HD + 2 * i;
  const size_t ob = ((size_t)h * SEQ + s) * HD + 2 * i;
  {
    float x1 = (float)qkv[qb], x2 = (float)qkv[qb + 1];
    qh[ob]     = (__bf16)(x1 * c - x2 * sn);
    qh[ob + 1] = (__bf16)(x1 * sn + x2 * c);
  }
  {
    float x1 = (float)qkv[qb + DM], x2 = (float)qkv[qb + DM + 1];
    kh[ob]     = (__bf16)(x1 * c - x2 * sn);
    kh[ob + 1] = (__bf16)(x1 * sn + x2 * c);
  }
}

// V -> vt[h][d][s]  (so PV B-fragments are contiguous per lane)
__global__ void v_transpose(const __bf16* __restrict__ qkv, __bf16* __restrict__ vt) {
  int idx = blockIdx.x * blockDim.x + threadIdx.x;
  if (idx >= SEQ * DM) return;
  const int d = idx & 63;
  const int h = (idx >> 6) & 15;
  const int s = idx >> 10;
  vt[((size_t)h * HD + d) * SEQ + s] = qkv[(size_t)s * QKV_N + 2 * DM + h * HD + d];
}

// ---------------------------------------------------------------------------
// Causal flash attention. Block = (head, 64 query rows); wave owns 16 rows.
// K/V tiles staged in LDS via double-buffered async global->LDS copies
// (shared by all 4 waves). Online softmax; P relayout D->A via per-wave LDS.
// ---------------------------------------------------------------------------
__global__ void __launch_bounds__(128)
flash_attn(const __bf16* __restrict__ qh, const __bf16* __restrict__ kh,
           const __bf16* __restrict__ vt, __bf16* __restrict__ ab) {
  __shared__ __bf16 lds_k[2][64][HD];  // 16 KB: K tile  [key][dim], double-buffered
  __shared__ __bf16 lds_v[2][HD][64];  // 16 KB: V tile  [dim][key], double-buffered
  __shared__ __bf16 lds_p[4][16][HD];  //  8 KB: per-wave 16x64 P tile

  const int tid    = threadIdx.x;
  const int lane   = tid & 31;
  const int wave   = tid >> 5;
  const int h      = blockIdx.x >> 6;
  const int qblock = blockIdx.x & 63;
  const int qbase  = qblock * 64 + wave * 16;
  const int halfsel = lane >> 4, col = lane & 15;

  const __bf16* Q  = qh + (size_t)h * SEQ * HD;
  const __bf16* Kh = kh + (size_t)h * SEQ * HD;
  const __bf16* Vt = vt + (size_t)h * HD * SEQ;

  // workgroup-relative LDS byte bases (LDS aperture: flat addr[31:0] = LDS offset)
  const unsigned ldsK0 = (unsigned)(uintptr_t)(void*)&lds_k[0][0][0];
  const unsigned ldsV0 = (unsigned)(uintptr_t)(void*)&lds_v[0][0][0];

  // Issue one 64x64 K tile + 64x64 V tile as async b128 copies (8 ops/thread).
  auto issue_tile = [&](int kv, int buf) {
    const unsigned kb = ldsK0 + (unsigned)buf * 64u * HD * 2u;
    const unsigned vb = ldsV0 + (unsigned)buf * 64u * HD * 2u;
#pragma unroll
    for (int c = 0; c < 4; ++c) {
      const int chunk = tid + c * 128;        // 0..511
      const int row   = chunk >> 3;           // 0..63
      const int col8  = (chunk & 7) * 8;      // bf16 elements (16B granularity)
      async_copy_b128(kb + (unsigned)(row * HD + col8) * 2u,
                      Kh + (size_t)(kv + row) * HD + col8);          // K[key][dim]
      async_copy_b128(vb + (unsigned)(row * 64 + col8) * 2u,
                      Vt + (size_t)row * SEQ + kv + col8);           // V^T[dim][key]
    }
  };

  v16bf aq[2];
#pragma unroll
  for (int ks = 0; ks < 2; ++ks)
    aq[ks] = load_a_frag(Q + (size_t)qbase * HD + ks * 32, HD, lane);

  v8f o[4] = {};
  float mrow[8], lrow[8];
#pragma unroll
  for (int r = 0; r < 8; ++r) { mrow[r] = -1e30f; lrow[r] = 0.0f; }

  const int ntiles = qblock + 1;  // causal: only tiles up to the diagonal
  issue_tile(0, 0);

  for (int t = 0; t < ntiles; ++t) {
    const int kv  = t * 64;
    const int cur = t & 1;

    wait_asynccnt0();    // my tile-t copies landed in LDS
    __syncthreads();     // everyone's copies landed; prior readers of buf 1-cur done

    if (t + 1 < ntiles) issue_tile(kv + 64, 1 - cur);

    const __bf16* Kt = &lds_k[cur][0][0];
    const __bf16* Vc = &lds_v[cur][0][0];

    // S = Q * K^T  (16 rows x 64 keys; 8 WMMA, B-frags from LDS)
    v8f s[4] = {};
#pragma unroll
    for (int ks = 0; ks < 2; ++ks)
#pragma unroll
      for (int f = 0; f < 4; ++f) {
        v16bf bk = load_b_frag(Kt + (16 * f) * HD + ks * 32, HD, lane);
        s[f] = __builtin_amdgcn_wmma_f32_16x16x32_bf16(
            false, aq[ks], false, bk, (short)0, s[f], false, false);
      }

    const bool last = (t == ntiles - 1);
#pragma unroll
    for (int r = 0; r < 8; ++r) {
      const int qrow = qbase + r + 8 * halfsel;
      float v[4], mx = -1e30f;
#pragma unroll
      for (int f = 0; f < 4; ++f) {
        float sv = s[f][r] * 0.125f;  // 1/sqrt(64)
        if (last && (kv + 16 * f + col) > qrow) sv = -1e30f;  // causal mask
        v[f] = sv;
        mx = fmaxf(mx, sv);
      }
#pragma unroll
      for (int d = 1; d < 16; d <<= 1) mx = fmaxf(mx, __shfl_xor(mx, d, 32));
      const float mn    = fmaxf(mrow[r], mx);
      const float alpha = __expf(mrow[r] - mn);
      float rs = 0.0f;
#pragma unroll
      for (int f = 0; f < 4; ++f) {
        float p = __expf(v[f] - mn);
        rs += p;
        lds_p[wave][r + 8 * halfsel][16 * f + col] = (__bf16)p;  // D-layout store
      }
#pragma unroll
      for (int d = 1; d < 16; d <<= 1) rs += __shfl_xor(rs, d, 32);
      lrow[r] = lrow[r] * alpha + rs;
      mrow[r] = mn;
#pragma unroll
      for (int f = 0; f < 4; ++f) o[f][r] *= alpha;
    }

    // O += P * V  (A-frags: per-wave LDS P; B-frags: shared LDS V tile; 8 WMMA)
#pragma unroll
    for (int ks = 0; ks < 2; ++ks) {
      v16bf ap = load_a_frag((const __bf16*)&lds_p[wave][0][0] + ks * 32, HD, lane);
#pragma unroll
      for (int f = 0; f < 4; ++f) {
        v16bf bv = load_b_frag(Vc + (16 * f) * 64 + ks * 32, 64, lane);
        o[f] = __builtin_amdgcn_wmma_f32_16x16x32_bf16(
            false, ap, false, bv, (short)0, o[f], false, false);
      }
    }
  }

  // normalize and scatter back to [s][h*64+d]
#pragma unroll
  for (int f = 0; f < 4; ++f)
#pragma unroll
    for (int r = 0; r < 8; ++r) {
      const int qrow = qbase + r + 8 * halfsel;
      ab[(size_t)qrow * DM + h * HD + 16 * f + col] = (__bf16)(o[f][r] / lrow[r]);
    }
}

// ---------------------------------------------------------------------------
// Launch: convert -> QKV GEMM -> RoPE/split + V^T -> flash attn -> out GEMM
// ---------------------------------------------------------------------------
extern "C" void kernel_launch(void* const* d_in, const int* in_sizes, int n_in,
                              void* d_out, int out_size, void* d_ws, size_t ws_size,
                              hipStream_t stream) {
  (void)in_sizes; (void)n_in; (void)out_size; (void)ws_size;
  const float* x     = (const float*)d_in[0];
  const int*   pos   = (const int*)d_in[1];
  const float* w_qkv = (const float*)d_in[2];
  const float* w_o   = (const float*)d_in[3];
  float* out = (float*)d_out;

  char* ws = (char*)d_ws;
  const size_t MiB = 1ull << 20;
  __bf16* xb    = (__bf16*)(ws + 0);        //  8 MiB: x bf16        (4096x1024)
  __bf16* wqkvb = (__bf16*)(ws + 8 * MiB);  //  6 MiB: w_qkv bf16    (3072x1024)
  __bf16* wob   = (__bf16*)(ws + 14 * MiB); //  2 MiB: w_o bf16      (1024x1024)
  __bf16* qkvb  = (__bf16*)(ws + 16 * MiB); // 24 MiB: qkv bf16      (4096x3072)
  __bf16* qhb   = (__bf16*)(ws + 40 * MiB); //  8 MiB: Q per-head    [16][4096][64]
  __bf16* khb   = (__bf16*)(ws + 48 * MiB); //  8 MiB: K per-head
  __bf16* vtb   = (__bf16*)(ws + 56 * MiB); //  8 MiB: V^T per-head  [16][64][4096]
  __bf16* abuf  = (__bf16*)(ws + 64 * MiB); //  8 MiB: attn out bf16 (4096x1024)

  f32_to_bf16<<<(SEQ * DM + 255) / 256, 256, 0, stream>>>(x, xb, SEQ * DM);
  f32_to_bf16<<<(QKV_N * DM + 255) / 256, 256, 0, stream>>>(w_qkv, wqkvb, QKV_N * DM);
  f32_to_bf16<<<(DM * DM + 255) / 256, 256, 0, stream>>>(w_o, wob, DM * DM);

  gemm_bf16<false><<<(SEQ / 64) * (QKV_N / 64) / 4, 128, 0, stream>>>(
      xb, wqkvb, (void*)qkvb, SEQ, QKV_N, DM);

  rope_split<<<(SEQ * NH * (HD / 2) + 255) / 256, 256, 0, stream>>>(qkvb, pos, qhb, khb);
  v_transpose<<<(SEQ * DM + 255) / 256, 256, 0, stream>>>(qkvb, vtb);

  flash_attn<<<NH * (SEQ / 64), 128, 0, stream>>>(qhb, khb, vtb, abuf);

  gemm_bf16<true><<<(SEQ / 64) * (DM / 64) / 4, 128, 0, stream>>>(
      abuf, wob, (void*)out, SEQ, DM, DM);
}
